// CrossAttention_37426345018109
// MI455X (gfx1250) — matrix operand order repaired and checked by reading, not compiled, not run on hardware
//
#include <hip/hip_runtime.h>
#include <hip/hip_bf16.h>

// ---------------- problem constants ----------------
#define BATCH 8
#define NSEQ  4096
#define CDIM  768
#define NH    12
#define DH    64

// ---------------- vector types ----------------
typedef __attribute__((ext_vector_type(16))) __bf16 bf16x16;
typedef __attribute__((ext_vector_type(8)))  __bf16 bf16x8;
typedef __attribute__((ext_vector_type(4)))  __bf16 bf16x4;
typedef __attribute__((ext_vector_type(8)))  float  f32x8;
typedef __attribute__((ext_vector_type(4)))  unsigned int u32x4;
typedef __attribute__((ext_vector_type(4)))  int    i32x4;
typedef __attribute__((ext_vector_type(8)))  int    i32x8;

static __device__ __forceinline__ f32x8 wmma_bf16(bf16x16 a, bf16x16 b, f32x8 c) {
  // D = A(16x32) * B(32x16) + C, f32 accumulate
  return __builtin_amdgcn_wmma_f32_16x16x32_bf16(false, a, false, b,
                                                 (short)0, c, false, false);
}

// A-fragment (16x32 bf16), ISA 7.12.2 layout:
//   lane<16 : M=lane,    K in [0..7]  and [16..23]
//   lane>=16: M=lane-16, K in [8..15] and [24..31]
static __device__ __forceinline__ bf16x16
load_frag_a(const __bf16* base, int stride, int m0, int k0, int lane) {
  int m  = m0 + (lane & 15);
  int kb = (lane >> 4) << 3;           // 0 or 8
  const __bf16* p = base + m * stride + k0 + kb;
  bf16x8 lo = *(const bf16x8*)p;        // K k0+kb   .. +7
  bf16x8 hi = *(const bf16x8*)(p + 16); // K k0+kb+16.. +23
  return __builtin_shufflevector(lo, hi, 0,1,2,3,4,5,6,7,8,9,10,11,12,13,14,15);
}

// B-fragment (32x16 bf16) from COLUMN-MAJOR storage T[n][k] (k contiguous):
//   lane<16 : N=lane,    K in [0..7]  and [16..23]
//   lane>=16: N=lane-16, K in [8..15] and [24..31]
static __device__ __forceinline__ bf16x16
load_frag_b(const __bf16* baseT, int stride, int n0, int k0, int lane) {
  int n  = n0 + (lane & 15);
  int kb = (lane >> 4) << 3;           // 0 or 8
  const __bf16* p = baseT + n * stride + k0 + kb;
  bf16x8 lo = *(const bf16x8*)p;
  bf16x8 hi = *(const bf16x8*)(p + 16);
  return __builtin_shufflevector(lo, hi, 0,1,2,3,4,5,6,7,8,9,10,11,12,13,14,15);
}

// ---- Tensor Data Mover: load a 64x64 bf16 tile (contiguous in global) into
// LDS with hardware row padding: +4 DWORDs after every 32 DWORDs (128B row),
// reproducing a 72-element padded row stride for conflict-free B-fragments.
// D# layout per CDNA5 ISA.
static __device__ __forceinline__ void tdm_load_ctx_tile(const void* gsrc, void* lds_dst) {
  unsigned long long ga = (unsigned long long)(uintptr_t)gsrc;
  unsigned int ldsa = (unsigned int)(uintptr_t)lds_dst; // low 32b = LDS byte offset
  u32x4 g0 = { 1u,                                   // count=1, user desc
               ldsa,                                  // lds_addr
               (unsigned int)ga,                      // global_addr[31:0]
               ((unsigned int)(ga >> 32) & 0x01FFFFFFu) | (2u << 30) }; // [56:32]|type=2
  i32x8 g1;
  g1[0] = 0x07110000;   // data_size=2B | pad_enable | pad_interval=32dw | pad_amount=4dw
  g1[1] = 64 << 16;     // tensor_dim0 = 64 (low 16 at bits 63:48)
  g1[2] = 64 << 16;     // tensor_dim1 = 64 (low 16 at bits 111:80 -> dword2[31:16])
  g1[3] = 64 << 16;     // tile_dim0 = 64 (bits 127:112)
  g1[4] = 64;           // tile_dim1 = 64, tile_dim2 = 0
  g1[5] = 64;           // tensor_dim0_stride = 64 (low 32)
  g1[6] = 64 << 16;     // tensor_dim1_stride = 64 (low 16 at bits 223:208)
  g1[7] = 0;
  i32x4 gz = {0, 0, 0, 0};
#if defined(__clang_major__) && (__clang_major__ >= 23)
  i32x8 gz8 = {0, 0, 0, 0, 0, 0, 0, 0};
  __builtin_amdgcn_tensor_load_to_lds(g0, g1, gz, gz, gz8, 0);
#else
  __builtin_amdgcn_tensor_load_to_lds(g0, g1, gz, gz, 0);
#endif
}

// ============================================================================
// Kernel 0: W_kv f32 -> bf16 (once; 2.25MB, L2 resident, read by all K1 blocks)
// ============================================================================
__global__ __launch_bounds__(256)
void convert_w_kernel(const float* __restrict__ W, unsigned short* __restrict__ Wb) {
  size_t i = ((size_t)blockIdx.x * 256 + threadIdx.x) * 4;
  float4 v = *(const float4*)(W + i);
  bf16x4 p = { (__bf16)v.x, (__bf16)v.y, (__bf16)v.z, (__bf16)v.w };
  *(bf16x4*)((__bf16*)Wb + i) = p;
}

// ============================================================================
// Kernel 1: fused kv-projection + ctx accumulation.
//   per block: 128 rows of segfeature, all 768 K, all 12 heads.
//   ctx_raw[b][h][d][e] += sum_m k[m][d]*v[m][e]   (f32 atomics, L2 resident)
// ============================================================================
#define MT1   128
#define SEGS  (CDIM + 8)   // 776 elems: 16B-aligned rows, conflict-free frags
#define KTS   (MT1 + 8)    // 136 elems
#define WSTR  40           // 32 K + pad; 80B rows (16B aligned)

__global__ __launch_bounds__(256)
void kv_ctx_kernel(const float* __restrict__ seg,
                   const unsigned short* __restrict__ Wb,
                   float* __restrict__ ctx_raw) {
  __shared__ __bf16 sseg[MT1 * SEGS];   // 198,656 B
  __shared__ __bf16 skT[DH * KTS];      //  17,408 B  (k transposed [d][m])
  __shared__ __bf16 svT[DH * KTS];      //  17,408 B  (v transposed [e][m])
  __shared__ __bf16 sWk[DH * WSTR];     //   5,120 B  (W k-cols, col-major [n][k])
  __shared__ __bf16 sWv[DH * WSTR];     //   5,120 B

  const int tid  = threadIdx.x;
  const int lane = tid & 31;
  const int w    = tid >> 5;            // 8 waves
  const int n0   = blockIdx.x * MT1;
  const int b    = blockIdx.y;
  const int m0   = w * 16;

  // ---- stage segfeature tile, f32 -> bf16 ----
  {
    const float4* g = (const float4*)(seg + ((size_t)b * NSEQ + n0) * CDIM);
    for (int idx = tid; idx < MT1 * (CDIM / 4); idx += 256) {
      int row = idx / (CDIM / 4), c4 = idx % (CDIM / 4);
      float4 v = g[(size_t)row * (CDIM / 4) + c4];
      bf16x4 p = { (__bf16)v.x, (__bf16)v.y, (__bf16)v.z, (__bf16)v.w };
      *(bf16x4*)&sseg[row * SEGS + c4 * 4] = p;
    }
  }

  const unsigned int* gw = (const unsigned int*)Wb;   // bf16 pairs

  for (int h = 0; h < NH; ++h) {
    f32x8 ka[4] = {}, va[4] = {};

    for (int kc = 0; kc < CDIM; kc += 32) {
      __syncthreads();   // prior chunk's fragments consumed (also covers sseg)
      // stage W chunk (bf16): rows kc..kc+31, cols h*64 (k) and C+h*64 (v),
      // transposed into column-major [n][k] LDS. 2048 dwords, coalesced.
      for (int idx = tid; idx < 2048; idx += 256) {
        int half = idx >> 10;
        int rem  = idx & 1023;
        int r = rem >> 5, cp = rem & 31;           // c = 2*cp, 2*cp+1
        unsigned int u =
            gw[(((size_t)(kc + r) * (2 * CDIM)) + half * CDIM + h * DH) / 2 + cp];
        unsigned short* sW = (unsigned short*)(half ? sWv : sWk);
        sW[(2 * cp) * WSTR + r]     = (unsigned short)(u & 0xFFFFu);
        sW[(2 * cp + 1) * WSTR + r] = (unsigned short)(u >> 16);
      }
      if (kc + 32 < CDIM) {  // hint next chunk toward L2 (global_prefetch_b8)
        __builtin_prefetch(
            (const char*)(gw + (((size_t)(kc + 32 + (tid & 31)) * (2 * CDIM)) + h * DH) / 2),
            0, 1);
      }
      __syncthreads();

      bf16x16 a = load_frag_a(sseg, SEGS, m0, kc, lane);
      #pragma unroll
      for (int nt = 0; nt < 4; ++nt) {
        bf16x16 bk = load_frag_b(sWk, WSTR, nt * 16, 0, lane);
        bf16x16 bv = load_frag_b(sWv, WSTR, nt * 16, 0, lane);
        ka[nt] = wmma_bf16(a, bk, ka[nt]);
        va[nt] = wmma_bf16(a, bv, va[nt]);
      }
    }

    // ---- spill k,v tiles transposed into LDS (bf16) ----
    // C/D layout: vgpr r -> M = m0 + ((lane>>4)<<3) + r, N = (lane&15)+16*nt
    __syncthreads();
    {
      int mrow = m0 + ((lane >> 4) << 3);
      #pragma unroll
      for (int nt = 0; nt < 4; ++nt) {
        int n = nt * 16 + (lane & 15);
        bf16x8 pk, pv;
        #pragma unroll
        for (int r = 0; r < 8; ++r) { pk[r] = (__bf16)ka[nt][r]; pv[r] = (__bf16)va[nt][r]; }
        *(bf16x8*)&skT[n * KTS + mrow] = pk;   // 16B contiguous store
        *(bf16x8*)&svT[n * KTS + mrow] = pv;
      }
    }
    __syncthreads();

    // ---- GEMM2: ctx[d][e] += sum_m kT[d][m] * v[m][e], K = 128 ----
    // 16 output tiles (4x4) over 8 waves -> 2 tiles each
    {
      int dt  = w >> 1;
      int et0 = (w & 1) * 2;
      f32x8 c0 = {}, c1 = {};
      #pragma unroll
      for (int km = 0; km < MT1; km += 32) {
        bf16x16 a2 = load_frag_a(skT, KTS, dt * 16, km, lane);
        bf16x16 b0 = load_frag_b(svT, KTS, (et0 + 0) * 16, km, lane);
        bf16x16 b1 = load_frag_b(svT, KTS, (et0 + 1) * 16, km, lane);
        c0 = wmma_bf16(a2, b0, c0);
        c1 = wmma_bf16(a2, b1, c1);
      }
      float* dst = ctx_raw + (size_t)(b * NH + h) * (DH * DH);
      int di = dt * 16 + ((lane >> 4) << 3);
      int e0 = (et0 + 0) * 16 + (lane & 15);
      int e1 = (et0 + 1) * 16 + (lane & 15);
      #pragma unroll
      for (int r = 0; r < 8; ++r) {
        atomicAdd(&dst[(di + r) * DH + e0], c0[r]);
        atomicAdd(&dst[(di + r) * DH + e1], c1[r]);
      }
    }
  }
}

// ============================================================================
// Kernel 2: ctx = softmax(scale * ctx, axis=d). Writes ctx TRANSPOSED [e][d]
// as bf16 so kernel 3 can read B-fragments contiguously.
// ============================================================================
__global__ __launch_bounds__(64)
void softmax_ctx_kernel(const float* __restrict__ ctx_raw,
                        unsigned short* __restrict__ ctxT_us) {
  __bf16* ctxT = (__bf16*)ctxT_us;
  const int bh = blockIdx.x;
  const int e  = threadIdx.x;
  const float scale = 0.125f;   // d^-0.5, d = 64
  const float* src = ctx_raw + (size_t)bh * DH * DH;

  float v[DH];
  float mx = -3.402823466e38f;
  #pragma unroll
  for (int d = 0; d < DH; ++d) {
    float t = src[d * DH + e] * scale;
    v[d] = t;
    mx = fmaxf(mx, t);
  }
  float s = 0.f;
  #pragma unroll
  for (int d = 0; d < DH; ++d) { float ex = __expf(v[d] - mx); v[d] = ex; s += ex; }
  float inv = 1.0f / s;
  __bf16* dst = ctxT + (size_t)bh * DH * DH + (size_t)e * DH;
  #pragma unroll
  for (int d = 0; d < DH; ++d) dst[d] = (__bf16)(v[d] * inv);
}

// ============================================================================
// Kernel 3: o[n][h*64+e] = sum_d x[n][h*64+d] * ctx[h][d][e]
// blockIdx.z selects x1/x2. 64-row tiles, 4 waves. ctx^T tile staged by the
// Tensor Data Mover (with hardware LDS row padding), waves overlap on WMMA.
// ============================================================================
#define MT3  64
#define XSTR (CDIM + 8)   // 776
#define CSTR 72           // 64 K + 4dw TDM pad; 144B rows (16B aligned)

__global__ __launch_bounds__(128)
void apply_ctx_kernel(const float* __restrict__ x1,
                      const float* __restrict__ x2,
                      const unsigned short* __restrict__ ctxT_us,
                      float* __restrict__ out_base) {
  __shared__ __bf16 sx[MT3 * XSTR];    // 99,328 B
  __shared__ __bf16 sct[DH * CSTR];    //  9,216 B  (ctx^T [e][d], TDM-padded)
  const __bf16* ctxT = (const __bf16*)ctxT_us;

  const int tid  = threadIdx.x;
  const int lane = tid & 31;
  const int w    = tid >> 5;           // 4 waves
  const int n0   = blockIdx.x * MT3;
  const int b    = blockIdx.y;
  const int src  = blockIdx.z;
  const float* x = src ? x2 : x1;
  float* out = out_base + (size_t)src * ((size_t)BATCH * NSEQ * CDIM);
  const int m0 = w * 16;

  // ---- stage x tile, f32 -> bf16 ----
  {
    const float4* g = (const float4*)(x + ((size_t)b * NSEQ + n0) * CDIM);
    for (int idx = tid; idx < MT3 * (CDIM / 4); idx += 128) {
      int row = idx / (CDIM / 4), c4 = idx % (CDIM / 4);
      float4 v = g[(size_t)row * (CDIM / 4) + c4];
      bf16x4 p = { (__bf16)v.x, (__bf16)v.y, (__bf16)v.z, (__bf16)v.w };
      *(bf16x4*)&sx[row * XSTR + c4 * 4] = p;
    }
  }

  for (int h = 0; h < NH; ++h) {
    __syncthreads();  // prior head's sct fragments consumed (also covers sx)
    // ---- TDM: DMA this head's ctx^T (64x64 bf16) into padded LDS rows ----
    if (w == 0) {
      tdm_load_ctx_tile(ctxT + (size_t)(b * NH + h) * DH * DH, &sct[0]);
      __builtin_amdgcn_s_wait_tensorcnt(0);
    }
    __syncthreads();

    f32x8 acc[4] = {};
    #pragma unroll
    for (int kd = 0; kd < DH; kd += 32) {
      bf16x16 a = load_frag_a(sx, XSTR, m0, h * DH + kd, lane);
      #pragma unroll
      for (int nt = 0; nt < 4; ++nt) {
        bf16x16 bb = load_frag_b(sct, CSTR, nt * 16, kd, lane);
        acc[nt] = wmma_bf16(a, bb, acc[nt]);
      }
    }

    int mrow = n0 + m0 + ((lane >> 4) << 3);
    #pragma unroll
    for (int nt = 0; nt < 4; ++nt) {
      int col = h * DH + nt * 16 + (lane & 15);
      #pragma unroll
      for (int r = 0; r < 8; ++r) {
        out[((size_t)b * NSEQ + mrow + r) * CDIM + col] = acc[nt][r];
      }
    }
  }
}

// ============================================================================
#define CTXRAW_BYTES ((size_t)BATCH * NH * DH * DH * sizeof(float))    // 1.5 MB
#define CTXT_BYTES   ((size_t)BATCH * NH * DH * DH * sizeof(short))    // 0.75 MB
#define WB_OFFSET    (CTXRAW_BYTES + CTXT_BYTES)
#define W_ELEMS      ((size_t)CDIM * 2 * CDIM)                         // 1,179,648

extern "C" void kernel_launch(void* const* d_in, const int* in_sizes, int n_in,
                              void* d_out, int out_size, void* d_ws, size_t ws_size,
                              hipStream_t stream) {
  const float* x1  = (const float*)d_in[0];
  const float* x2  = (const float*)d_in[1];
  const float* seg = (const float*)d_in[2];
  const float* Wkv = (const float*)d_in[3];
  float* out = (float*)d_out;

  float*          ctx_raw = (float*)d_ws;
  unsigned short* ctxT    = (unsigned short*)((char*)d_ws + CTXRAW_BYTES);
  unsigned short* Wb      = (unsigned short*)((char*)d_ws + WB_OFFSET);

  hipMemsetAsync(ctx_raw, 0, CTXRAW_BYTES, stream);
  convert_w_kernel<<<dim3(W_ELEMS / (256 * 4)), 256, 0, stream>>>(Wkv, Wb);

  kv_ctx_kernel<<<dim3(NSEQ / MT1, BATCH), 256, 0, stream>>>(seg, Wb, ctx_raw);
  softmax_ctx_kernel<<<dim3(BATCH * NH), 64, 0, stream>>>(ctx_raw, ctxT);
  apply_ctx_kernel<<<dim3(NSEQ / MT3, BATCH, 2), 128, 0, stream>>>(x1, x2, ctxT, out);
}
